// TripleGNNFeatureExtractor_12575664243128
// MI455X (gfx1250) — matrix operand order repaired and checked by reading, not compiled, not run on hardware
//
#include <hip/hip_runtime.h>
#include <hip/hip_bf16.h>

typedef __attribute__((ext_vector_type(16))) _Float16 v16h;
typedef __attribute__((ext_vector_type(8)))  float    v8f;
typedef __attribute__((ext_vector_type(4)))  float    v4f;

#define MTILES 4   // 16-row tiles per block: amortizes LDS-staged W across 64 rows

// ---------------------------------------------------------------- utilities
__global__ void k_zero(float* __restrict__ p, long n) {
    long i = (long)blockIdx.x * blockDim.x + threadIdx.x;
    long stride = (long)gridDim.x * blockDim.x;
    for (; i < n; i += stride) p[i] = 0.0f;
}

__global__ void k_degree(const int* __restrict__ dst, float* __restrict__ deg, int E) {
    int e = blockIdx.x * blockDim.x + threadIdx.x;
    if (e < E) atomicAdd(&deg[dst[e]], 1.0f);
}

__global__ void k_rsqrt(const float* __restrict__ deg, float* __restrict__ dinv, int n) {
    int i = blockIdx.x * blockDim.x + threadIdx.x;
    if (i < n) dinv[i] = rsqrtf(deg[i] + 1.0f);
}

// ---------------------------------------------------------------- GEMM (WMMA)
// C[N,128] = A[N,K] @ W[K,128]   (f16 WMMA, f32 accumulate)
// Block = 8 waves; wave w owns column tile w; block covers MTILES row tiles.
// W staged once per block into LDS, pre-swizzled into B-fragment layout:
//   frag(ks,ct): lane l holds col = ct*16+(l&15), K = ks*32+(l>>4)*16 + j, j=0..15
// Loop order: K-step outer, B fragment loaded once, 4 WMMAs back-to-back.
__global__ __launch_bounds__(256)
void k_gemm(const float* __restrict__ A, const float* __restrict__ W,
            float* __restrict__ C, int N, int K) {
    __shared__ _Float16 Wlds[256 * 128];   // 64 KB; K <= 256

    const int tid = threadIdx.x;
    for (int idx = tid; idx < K * 128; idx += 256) {
        int k   = idx >> 7;
        int col = idx & 127;
        int ks  = k >> 5;
        int kr  = k & 31;
        int lane = ((kr >> 4) << 4) | (col & 15);
        int j    = kr & 15;
        int ct   = col >> 4;
        Wlds[((((ks << 3) + ct) << 5) + lane) * 16 + j] = (_Float16)W[idx];
    }
    __syncthreads();

    const int wave   = tid >> 5;           // column tile 0..7
    const int lane   = tid & 31;
    const int tileM0 = blockIdx.x * MTILES;
    const int kbase  = (lane >> 4) << 4;   // 0 or 16 (A-layout hi-lane K group)

    const float* aptr[MTILES];
#pragma unroll
    for (int mt = 0; mt < MTILES; ++mt) {
        int row  = (tileM0 + mt) * 16 + (lane & 15);
        int rowA = row < N ? row : N - 1;              // clamp loads; stores predicated
        aptr[mt] = A + (long)rowA * K;
    }

    v8f c[MTILES] = {};
    const int nsteps = K >> 5;
    for (int ks = 0; ks < nsteps; ++ks) {
        const int k0 = ks << 5;
        const v16h b = *(const v16h*)&Wlds[((((ks << 3) + wave) << 5) + lane) * 16];
#pragma unroll
        for (int mt = 0; mt < MTILES; ++mt) {
            const float* ap = aptr[mt] + k0 + kbase;
            v4f a0 = *(const v4f*)(ap);         // K = k0+kbase+0..3
            v4f a1 = *(const v4f*)(ap + 4);     // K = k0+kbase+4..7
            v4f a2 = *(const v4f*)(ap + 8);     // K = k0+kbase+8..11
            v4f a3 = *(const v4f*)(ap + 12);    // K = k0+kbase+12..15
            v16h a;
#pragma unroll
            for (int t = 0; t < 4; ++t) {
                a[t]      = (_Float16)a0[t];
                a[4 + t]  = (_Float16)a1[t];
                a[8 + t]  = (_Float16)a2[t];
                a[12 + t] = (_Float16)a3[t];
            }
            c[mt] = __builtin_amdgcn_wmma_f32_16x16x32_f16(
                        false, a, false, b, (short)0, c[mt], false, false);
        }
    }

    // C/D layout: VGPR r -> M = (lane>>4)*8 + r, N = lane&15
    const int colBase = wave * 16 + (lane & 15);
    const int rbase   = (lane >> 4) << 3;
#pragma unroll
    for (int mt = 0; mt < MTILES; ++mt) {
#pragma unroll
        for (int r = 0; r < 8; ++r) {
            int ro = (tileM0 + mt) * 16 + rbase + r;
            if (ro < N) C[(long)ro * 128 + colBase] = c[mt][r];
        }
    }
}

// ---------------------------------------------------------------- edge scatter
// one wave per edge; 32 lanes x 4 floats = 128 features
__global__ __launch_bounds__(256)
void k_edge_agg(const float* __restrict__ h, const float* __restrict__ dinv,
                const int* __restrict__ src, const int* __restrict__ dst,
                float* __restrict__ agg, int E) {
    int gid  = blockIdx.x * blockDim.x + threadIdx.x;
    int e    = gid >> 5;
    int lane = gid & 31;
    if (e >= E) return;
    int s = src[e], d = dst[e];
    float nrm = dinv[s] * dinv[d];
    const v4f hv = *(const v4f*)(h + (long)s * 128 + lane * 4);
    float* out = agg + (long)d * 128 + lane * 4;
    atomicAdd(out + 0, hv[0] * nrm);
    atomicAdd(out + 1, hv[1] * nrm);
    atomicAdd(out + 2, hv[2] * nrm);
    atomicAdd(out + 3, hv[3] * nrm);
}

// out = agg + h*dinv^2 + bias  (optional ReLU)
__global__ void k_finalize(const float* __restrict__ agg, const float* __restrict__ h,
                           const float* __restrict__ dinv, const float* __restrict__ bias,
                           float* __restrict__ out, int N, int relu) {
    long idx = (long)blockIdx.x * blockDim.x + threadIdx.x;
    if (idx >= (long)N * 128) return;
    int i = (int)(idx >> 7);
    int f = (int)(idx & 127);
    float di = dinv[i];
    float v = agg[idx] + h[idx] * di * di + bias[f];
    if (relu) v = v > 0.0f ? v : 0.0f;
    out[idx] = v;
}

// ---------------------------------------------------------------- attention
__global__ __launch_bounds__(256)
void k_attention(const float* __restrict__ f0, const float* __restrict__ f1,
                 const float* __restrict__ f2, const float* __restrict__ Wa,
                 const float* __restrict__ ba, float* __restrict__ out, int N) {
    int gid  = blockIdx.x * blockDim.x + threadIdx.x;
    int i    = gid >> 5;
    int lane = gid & 31;
    if (i >= N) return;
    long base = (long)i * 128 + lane * 4;
    v4f p  = *(const v4f*)(f0 + base);
    v4f q  = *(const v4f*)(f1 + base);
    v4f g  = *(const v4f*)(f2 + base);
    v4f wa = *(const v4f*)(Wa + lane * 4);
    float l0 = 0.f, l1 = 0.f, l2 = 0.f;
#pragma unroll
    for (int t = 0; t < 4; ++t) {
        l0 += p[t] * wa[t];
        l1 += q[t] * wa[t];
        l2 += g[t] * wa[t];
    }
    for (int off = 16; off > 0; off >>= 1) {   // wave32 reduction
        l0 += __shfl_xor(l0, off, 32);
        l1 += __shfl_xor(l1, off, 32);
        l2 += __shfl_xor(l2, off, 32);
    }
    float bb = ba[0];
    l0 += bb; l1 += bb; l2 += bb;
    float m  = fmaxf(l0, fmaxf(l1, l2));
    float e0 = __expf(l0 - m), e1 = __expf(l1 - m), e2 = __expf(l2 - m);
    float inv = 1.0f / (e0 + e1 + e2);
    float w0 = e0 * inv, w1 = e1 * inv, w2 = e2 * inv;
    v4f r;
#pragma unroll
    for (int t = 0; t < 4; ++t) r[t] = p[t] * w0 + q[t] * w1 + g[t] * w2;
    *(v4f*)(out + base) = r;
    if (lane == 0) {
        float* aw = out + (long)N * 128 + (long)i * 3;
        aw[0] = w0; aw[1] = w1; aw[2] = w2;
    }
}

// ---------------------------------------------------------------- launch
extern "C" void kernel_launch(void* const* d_in, const int* in_sizes, int n_in,
                              void* d_out, int out_size, void* d_ws, size_t ws_size,
                              hipStream_t stream) {
    const float* x = (const float*)d_in[0];
    const int* ei[3] = {(const int*)d_in[1], (const int*)d_in[2], (const int*)d_in[3]};
    const float* W1[3] = {(const float*)d_in[4],  (const float*)d_in[8],  (const float*)d_in[12]};
    const float* b1[3] = {(const float*)d_in[5],  (const float*)d_in[9],  (const float*)d_in[13]};
    const float* W2[3] = {(const float*)d_in[6],  (const float*)d_in[10], (const float*)d_in[14]};
    const float* b2[3] = {(const float*)d_in[7],  (const float*)d_in[11], (const float*)d_in[15]};
    const float* Wa = (const float*)d_in[16];
    const float* ba = (const float*)d_in[17];

    const int N = in_sizes[0] / 256;   // IN_DIM = 256
    const int E = in_sizes[1] / 2;     // edge_index is [2, E]
    const long NF = (long)N * 128;

    float* ws = (float*)d_ws;
    size_t off = 0;
    float* dinv = ws;             off += (size_t)3 * N;
    float* feat[3];
    for (int b = 0; b < 3; ++b) { feat[b] = ws + off; off += (size_t)NF; }
    float* tmpA = ws + off;       off += (size_t)NF;   // GEMM output h
    float* h1   = ws + off;       off += (size_t)NF;   // layer-1 activation
    float* tmpB = ws + off;       off += (size_t)NF;   // agg / deg scratch

    const dim3 blk(256);
    const int gN   = (N + 255) / 256;
    const int gNF  = (int)((NF + 255) / 256);
    const int gE   = (E + 255) / 256;
    const int gE32 = (int)(((long)E * 32 + 255) / 256);
    const int gN32 = (int)(((long)N * 32 + 255) / 256);
    const int gM   = (N + 16 * MTILES - 1) / (16 * MTILES);

    for (int b = 0; b < 3; ++b) {
        const int* src = ei[b];
        const int* dst = ei[b] + E;
        float* di = dinv + (size_t)b * N;

        // symmetric normalization terms
        k_zero  <<<gN,  blk, 0, stream>>>(tmpB, N);
        k_degree<<<gE,  blk, 0, stream>>>(dst, tmpB, E);
        k_rsqrt <<<gN,  blk, 0, stream>>>(tmpB, di, N);

        // layer 1: h = X @ W1 ; agg ; relu(agg + h*dinv^2 + b1)
        k_gemm    <<<gM,   blk, 0, stream>>>(x, W1[b], tmpA, N, 256);
        k_zero    <<<gNF,  blk, 0, stream>>>(tmpB, NF);
        k_edge_agg<<<gE32, blk, 0, stream>>>(tmpA, di, src, dst, tmpB, E);
        k_finalize<<<gNF,  blk, 0, stream>>>(tmpB, tmpA, di, b1[b], h1, N, 1);

        // layer 2
        k_gemm    <<<gM,   blk, 0, stream>>>(h1, W2[b], tmpA, N, 128);
        k_zero    <<<gNF,  blk, 0, stream>>>(tmpB, NF);
        k_edge_agg<<<gE32, blk, 0, stream>>>(tmpA, di, src, dst, tmpB, E);
        k_finalize<<<gNF,  blk, 0, stream>>>(tmpB, tmpA, di, b2[b], feat[b], N, 0);
    }

    k_attention<<<gN32, blk, 0, stream>>>(feat[0], feat[1], feat[2], Wa, ba,
                                          (float*)d_out, N);
}
